// MultiHeadAttention_20212116095568
// MI455X (gfx1250) — matrix operand order repaired and checked by reading, not compiled, not run on hardware
//
#include <hip/hip_runtime.h>

#define D_MODEL 768
#define N_HEADS 8
#define D_HEAD  96
#define SEQ     2048
#define BATCH   4
#define ROWS    (BATCH * SEQ)          // 8192
#define KT      (D_MODEL / 32)         // 24 k-chunks of 32
#define ATTN_SCALE 0.28867513459481287f  // 12^-0.5 (faithful to source)
#define LN_EPS 1e-5f

typedef __bf16 v16bf __attribute__((ext_vector_type(16)));
typedef float  v8f   __attribute__((ext_vector_type(8)));
typedef unsigned int v4u __attribute__((ext_vector_type(4)));
typedef __attribute__((ext_vector_type(4))) unsigned int u32x4;
typedef __attribute__((ext_vector_type(8))) int i32x8;
typedef __attribute__((ext_vector_type(4))) int i32x4;

union FragB { v16bf bf; v4u q[2]; };

__device__ __forceinline__ unsigned short f2bf(float f) {
  unsigned int u = __float_as_uint(f);
  u += 0x7FFFu + ((u >> 16) & 1u);   // round-to-nearest-even
  return (unsigned short)(u >> 16);
}

#define WMMA_BF16(a, b, c) \
  __builtin_amdgcn_wmma_f32_16x16x32_bf16(false, (a), false, (b), (short)0, (c), false, false)

// ---- Tensor Data Mover: 2D tile load (6 rows x 512 elems bf16, stride 12288 elems) ----
__device__ __forceinline__ void tdm_load_wtile(unsigned lds_off, const unsigned short* g) {
  unsigned long long ga = (unsigned long long)(size_t)g;
  u32x4 g0 = { 1u,                                    // count=1 (valid descriptor)
               lds_off,                               // lds_addr
               (unsigned)ga,                          // global_addr[31:0]
               ((unsigned)(ga >> 32) & 0x01FFFFFFu) | 0x80000000u };  // addr[56:32] | type=2
  i32x8 g1 = { (int)0x00010000,                      // data_size=1 (2 bytes), no multicast
               0,                                    // atomic barrier addr=0, dim0 lo16=0
               (int)(0x10u | (0x1000u << 16)),       // tensor_dim0=1<<20 hi, tensor_dim1=4096 lo
               (int)(512u << 16),                    // tensor_dim1 hi=0, tile_dim0=512
               6,                                    // tile_dim1=6, tile_dim2=0
               12288,                                // tensor_dim0_stride = 24*512 elems
               0, 0 };
  i32x4 z4 = {0, 0, 0, 0};
#if __clang_major__ >= 23
  i32x8 z8 = {0, 0, 0, 0, 0, 0, 0, 0};
  __builtin_amdgcn_tensor_load_to_lds(g0, g1, z4, z4, z8, 0);
#else
  __builtin_amdgcn_tensor_load_to_lds(g0, g1, z4, z4, 0);
#endif
}

// ---------------- fp32 -> bf16 activation convert ----------------
__global__ void cvt_bf16_kernel(const float* __restrict__ in,
                                unsigned short* __restrict__ out, int n) {
  int i = blockIdx.x * blockDim.x + threadIdx.x;
  if (i < n) out[i] = f2bf(in[i]);
}

// ------- weight fp32 [K=768][N=768] -> bf16 fragment-swizzled -------
// layout: [nt (48)][kt (24)][lane (32)][e (16)]
__global__ void swz_w_kernel(const float* __restrict__ W,
                             unsigned short* __restrict__ out) {
  int tid = blockIdx.x * blockDim.x + threadIdx.x;  // 768*768 threads
  int e    = tid & 15;
  int lane = (tid >> 4) & 31;
  int kt   = (tid >> 9) % KT;
  int nt   = (tid >> 9) / KT;
  int k = kt * 32 + ((lane >> 4) << 4) + e;
  int n = nt * 16 + (lane & 15);
  out[tid] = f2bf(W[k * D_MODEL + n]);
}

// ---------------- QKV projection: one wave per 16x96 tile ----------------
// TDM stages the 6 weight fragments per k-chunk into an LDS double buffer.
// mode 0: K -> [B,H,S,Dh]; mode 1: V -> [B,H,Dh,S]; mode 2: Q scaled -> [B,H,S,Dh]
__global__ __launch_bounds__(32) void proj_gemm_kernel(
    const unsigned short* __restrict__ Xbf,   // [8192][768] bf16
    const unsigned short* __restrict__ Wswz,  // swizzled bf16
    const float* __restrict__ bias,           // [768]
    unsigned short* __restrict__ outp, int mode) {
  __shared__ unsigned short wbuf[2][6 * 512];   // 12 KB double buffer
  int lane  = threadIdx.x;
  int m0    = blockIdx.x * 16;
  int nt0   = blockIdx.y * 6;
  int laneN = lane & 15;
  int half8 = (lane >> 4) << 3;
  int half16 = (lane >> 4) << 4;

  const unsigned short* wsrc = Wswz + (size_t)(nt0 * KT) * 512;
  tdm_load_wtile((unsigned)(size_t)&wbuf[0][0], wsrc);  // kc = 0

  const unsigned short* arow = Xbf + (size_t)(m0 + laneN) * D_MODEL;
  FragB a, an;
  a.q[0] = *(const v4u*)(arow + half16);
  a.q[1] = *(const v4u*)(arow + half16 + 8);

  v8f acc[6];
  v8f zero = {0.f, 0.f, 0.f, 0.f, 0.f, 0.f, 0.f, 0.f};
#pragma unroll
  for (int j = 0; j < 6; ++j) acc[j] = zero;

  for (int kc = 0; kc < KT; ++kc) {
    if (kc + 1 < KT) {
      tdm_load_wtile((unsigned)(size_t)&wbuf[(kc + 1) & 1][0],
                     wsrc + (size_t)(kc + 1) * 512);
      int d0 = (kc + 1) * 32 + half16;
      an.q[0] = *(const v4u*)(arow + d0);
      an.q[1] = *(const v4u*)(arow + d0 + 8);
      asm volatile("s_wait_tensorcnt 0x1" ::: "memory");  // wait for buf[kc]
    } else {
      asm volatile("s_wait_tensorcnt 0x0" ::: "memory");
    }
    const unsigned short* wb = &wbuf[kc & 1][0];
#pragma unroll
    for (int j = 0; j < 6; ++j) {
      FragB b;
      b.q[0] = *(const v4u*)(wb + (j * 32 + lane) * 16);
      b.q[1] = *(const v4u*)(wb + (j * 32 + lane) * 16 + 8);
      acc[j] = WMMA_BF16(a.bf, b.bf, acc[j]);
    }
    a = an;
  }

  int rbase = m0 + half8;
#pragma unroll
  for (int j = 0; j < 6; ++j) {
    int n = (nt0 + j) * 16 + laneN;
    float bv = bias[n];
    int h = n / D_HEAD, d = n % D_HEAD;
#pragma unroll
    for (int i = 0; i < 8; ++i) {
      int m = rbase + i;
      int b = m >> 11, s = m & (SEQ - 1);
      float v = acc[j][i] + bv;
      if (mode == 2) v *= ATTN_SCALE;
      int idx;
      if (mode == 1) idx = ((b * N_HEADS + h) * D_HEAD + d) * SEQ + s;
      else           idx = ((b * N_HEADS + h) * SEQ + s) * D_HEAD + d;
      outp[idx] = f2bf(v);
    }
  }
}

// ---------------- flash attention: one wave per 16-row q tile ----------------
__global__ __launch_bounds__(256) void flash_attn_kernel(
    const unsigned short* __restrict__ Qbf,  // [B,H,S,96] pre-scaled
    const unsigned short* __restrict__ Kbf,  // [B,H,S,96]
    const unsigned short* __restrict__ Vt,   // [B,H,96,S]
    unsigned short* __restrict__ ctx) {      // [8192][768] bf16
  __shared__ unsigned short Plds[8][16][64];
  int lane = threadIdx.x & 31;
  int w    = threadIdx.x >> 5;
  int qt   = blockIdx.x * 8 + w;           // 0..4095
  int st   = qt & (SEQ / 16 - 1);
  int bh   = qt >> 7;
  int q0   = st * 16;
  const unsigned short* Qb = Qbf + (size_t)bh * SEQ * D_HEAD;
  const unsigned short* Kb = Kbf + (size_t)bh * SEQ * D_HEAD;
  const unsigned short* Vb = Vt  + (size_t)bh * D_HEAD * SEQ;

  int laneN  = lane & 15;
  int half8  = (lane >> 4) << 3;
  int half16 = (lane >> 4) << 4;

  FragB aq[3];
  const unsigned short* qrow = Qb + (size_t)(q0 + laneN) * D_HEAD;
#pragma unroll
  for (int kc = 0; kc < 3; ++kc) {
    int d0 = kc * 32 + half16;
    aq[kc].q[0] = *(const v4u*)(qrow + d0);
    aq[kc].q[1] = *(const v4u*)(qrow + d0 + 8);
  }

  v8f zero = {0.f, 0.f, 0.f, 0.f, 0.f, 0.f, 0.f, 0.f};
  v8f cacc[6];
#pragma unroll
  for (int j = 0; j < 6; ++j) cacc[j] = zero;
  float mo[8], lo[8];
#pragma unroll
  for (int i = 0; i < 8; ++i) { mo[i] = -3.0e38f; lo[i] = 0.f; }

  for (int kv0 = 0; kv0 < SEQ; kv0 += 64) {
    // prefetch next kv block (K rows + V row segments)
    if (kv0 + 64 < SEQ) {
      const unsigned short* nk = Kb + (size_t)(kv0 + 64 + lane) * D_HEAD;
      __builtin_prefetch(nk, 0, 1);
      __builtin_prefetch(nk + 32 * D_HEAD, 0, 1);
      const unsigned short* nv = Vb + (size_t)lane * SEQ + kv0 + 64;
      __builtin_prefetch(nv, 0, 1);
      __builtin_prefetch(nv + (size_t)32 * SEQ, 0, 1);
      __builtin_prefetch(nv + (size_t)64 * SEQ, 0, 1);
    }

    // ---- scores: 16 q x 64 kv, K-fragments double-buffered across n-tiles ----
    FragB k0[3], k1[3];
    {
      const unsigned short* krow = Kb + (size_t)(kv0 + laneN) * D_HEAD;
#pragma unroll
      for (int kc = 0; kc < 3; ++kc) {
        k0[kc].q[0] = *(const v4u*)(krow + kc * 32 + half16);
        k0[kc].q[1] = *(const v4u*)(krow + kc * 32 + half16 + 8);
      }
    }
    v8f s[4];
#pragma unroll
    for (int nt = 0; nt < 4; ++nt) {
      FragB* cur = (nt & 1) ? k1 : k0;
      FragB* nxt = (nt & 1) ? k0 : k1;
      if (nt < 3) {
        const unsigned short* krow = Kb + (size_t)(kv0 + (nt + 1) * 16 + laneN) * D_HEAD;
#pragma unroll
        for (int kc = 0; kc < 3; ++kc) {
          nxt[kc].q[0] = *(const v4u*)(krow + kc * 32 + half16);
          nxt[kc].q[1] = *(const v4u*)(krow + kc * 32 + half16 + 8);
        }
      }
      s[nt] = zero;
#pragma unroll
      for (int kc = 0; kc < 3; ++kc)
        s[nt] = WMMA_BF16(aq[kc].bf, cur[kc].bf, s[nt]);
    }

    // ---- online softmax (row stats per lane-half, matching C-layout) ----
    float bm[8], al[8], rs[8];
#pragma unroll
    for (int i = 0; i < 8; ++i)
      bm[i] = fmaxf(fmaxf(s[0][i], s[1][i]), fmaxf(s[2][i], s[3][i]));
#pragma unroll
    for (int i = 0; i < 8; ++i) {
      for (int off = 1; off < 16; off <<= 1)
        bm[i] = fmaxf(bm[i], __shfl_xor(bm[i], off, 16));
      float mn = fmaxf(mo[i], bm[i]);
      al[i] = __expf(mo[i] - mn);
      mo[i] = mn;
    }
#pragma unroll
    for (int nt = 0; nt < 4; ++nt)
#pragma unroll
      for (int i = 0; i < 8; ++i)
        s[nt][i] = __expf(s[nt][i] - mo[i]);
#pragma unroll
    for (int i = 0; i < 8; ++i) {
      rs[i] = s[0][i] + s[1][i] + s[2][i] + s[3][i];
      for (int off = 1; off < 16; off <<= 1)
        rs[i] += __shfl_xor(rs[i], off, 16);
      lo[i] = lo[i] * al[i] + rs[i];
    }
#pragma unroll
    for (int j = 0; j < 6; ++j)
#pragma unroll
      for (int i = 0; i < 8; ++i) cacc[j][i] *= al[i];

    // ---- P -> LDS (bf16) for D-layout -> A-layout transpose ----
#pragma unroll
    for (int nt = 0; nt < 4; ++nt)
#pragma unroll
      for (int i = 0; i < 8; ++i)
        Plds[w][i + half8][nt * 16 + laneN] = f2bf(s[nt][i]);
    asm volatile("s_wait_dscnt 0x0" ::: "memory");

    // ---- ctx += P * V, V fragments pipelined ----
    FragB pa0, pa1;
    const unsigned short* prow = &Plds[w][laneN][0];
    pa0.q[0] = *(const v4u*)(prow + half16);
    pa0.q[1] = *(const v4u*)(prow + half16 + 8);
    pa1.q[0] = *(const v4u*)(prow + 32 + half16);
    pa1.q[1] = *(const v4u*)(prow + 32 + half16 + 8);

    FragB v0, v1;
    {
      const unsigned short* vrow = Vb + (size_t)laneN * SEQ + kv0 + half16;
      v0.q[0] = *(const v4u*)(vrow);
      v0.q[1] = *(const v4u*)(vrow + 8);
    }
#pragma unroll
    for (int t = 0; t < 12; ++t) {
      int c = t / 6, j = t % 6;
      FragB& cur = (t & 1) ? v1 : v0;
      FragB& nxt = (t & 1) ? v0 : v1;
      if (t < 11) {
        int tn = t + 1, cn = tn / 6, jn = tn % 6;
        const unsigned short* vrow =
            Vb + (size_t)(jn * 16 + laneN) * SEQ + kv0 + cn * 32 + half16;
        nxt.q[0] = *(const v4u*)(vrow);
        nxt.q[1] = *(const v4u*)(vrow + 8);
      }
      cacc[j] = WMMA_BF16((c ? pa1 : pa0).bf, cur.bf, cacc[j]);
    }
  }

  // normalize and store ctx (bf16, row-major [8192][768])
  int b = bh >> 3, h = bh & 7;
#pragma unroll
  for (int i = 0; i < 8; ++i) {
    float inv = 1.0f / lo[i];
    int m = b * SEQ + q0 + i + half8;
#pragma unroll
    for (int j = 0; j < 6; ++j) {
      int n = h * D_HEAD + j * 16 + laneN;
      ctx[(size_t)m * D_MODEL + n] = f2bf(cacc[j][i] * inv);
    }
  }
}

// ---------- out-proj + bias + residual + LayerNorm (fused) ----------
__global__ __launch_bounds__(256) void oproj_ln_kernel(
    const unsigned short* __restrict__ ctx,   // [8192][768] bf16
    const unsigned short* __restrict__ Wswz,  // swizzled Wo
    const float* __restrict__ bo,
    const float* __restrict__ residual,       // query_in fp32
    const float* __restrict__ gamma, const float* __restrict__ beta,
    float* __restrict__ out) {
  __shared__ float xs[16][D_MODEL];
  int lane = threadIdx.x & 31, w = threadIdx.x >> 5;
  int m0 = blockIdx.x * 16;
  int laneN = lane & 15;
  int half8 = (lane >> 4) << 3;
  int half16 = (lane >> 4) << 4;

  v8f zero = {0.f, 0.f, 0.f, 0.f, 0.f, 0.f, 0.f, 0.f};
  v8f acc[6];
#pragma unroll
  for (int j = 0; j < 6; ++j) acc[j] = zero;

  const unsigned short* arow  = ctx + (size_t)(m0 + laneN) * D_MODEL;
  const unsigned short* bbase = Wswz + (size_t)(w * 6) * KT * 512;
  // fragment address for (j, kc): bbase + ((j*KT + kc)*32 + lane)*16

  FragB a, an, bc, bn;
  a.q[0] = *(const v4u*)(arow + half16);
  a.q[1] = *(const v4u*)(arow + half16 + 8);
  bc.q[0] = *(const v4u*)(bbase + lane * 16);
  bc.q[1] = *(const v4u*)(bbase + lane * 16 + 8);

  for (int kc = 0; kc < KT; ++kc) {
#pragma unroll
    for (int j = 0; j < 6; ++j) {
      if (j < 5) {
        const unsigned short* bp = bbase + (size_t)((j + 1) * KT + kc) * 512 + lane * 16;
        bn.q[0] = *(const v4u*)(bp);
        bn.q[1] = *(const v4u*)(bp + 8);
      } else if (kc + 1 < KT) {
        const unsigned short* bp = bbase + (size_t)(kc + 1) * 512 + lane * 16;
        bn.q[0] = *(const v4u*)(bp);
        bn.q[1] = *(const v4u*)(bp + 8);
        int d0 = (kc + 1) * 32 + half16;
        an.q[0] = *(const v4u*)(arow + d0);
        an.q[1] = *(const v4u*)(arow + d0 + 8);
      }
      acc[j] = WMMA_BF16(a.bf, bc.bf, acc[j]);
      bc = bn;
    }
    a = an;
  }

#pragma unroll
  for (int j = 0; j < 6; ++j) {
    int n = (w * 6 + j) * 16 + laneN;
    float bv = bo[n];
#pragma unroll
    for (int i = 0; i < 8; ++i) {
      int r = i + half8;
      xs[r][n] = acc[j][i] + bv + residual[(size_t)(m0 + r) * D_MODEL + n];
    }
  }
  __syncthreads();

  // LayerNorm: wave w handles rows 2w and 2w+1
#pragma unroll
  for (int rr = 0; rr < 2; ++rr) {
    int r = w * 2 + rr;
    float sum = 0.f, sq = 0.f;
#pragma unroll
    for (int t = 0; t < D_MODEL / 32; ++t) {
      float v = xs[r][lane + t * 32];
      sum += v; sq += v * v;
    }
    for (int off = 1; off < 32; off <<= 1) {
      sum += __shfl_xor(sum, off, 32);
      sq  += __shfl_xor(sq, off, 32);
    }
    float mean = sum * (1.0f / D_MODEL);
    float var  = sq * (1.0f / D_MODEL) - mean * mean;
    float rstd = rsqrtf(var + LN_EPS);
#pragma unroll
    for (int t = 0; t < D_MODEL / 32; ++t) {
      int c = lane + t * 32;
      out[(size_t)(m0 + r) * D_MODEL + c] =
          (xs[r][c] - mean) * rstd * gamma[c] + beta[c];
    }
  }
}

extern "C" void kernel_launch(void* const* d_in, const int* in_sizes, int n_in,
                              void* d_out, int out_size, void* d_ws, size_t ws_size,
                              hipStream_t stream) {
  (void)in_sizes; (void)n_in; (void)out_size; (void)ws_size;
  const float* key_in   = (const float*)d_in[0];
  const float* value_in = (const float*)d_in[1];
  const float* query_in = (const float*)d_in[2];
  const float* Wk = (const float*)d_in[3];  const float* bk = (const float*)d_in[4];
  const float* Wv = (const float*)d_in[5];  const float* bv = (const float*)d_in[6];
  const float* Wq = (const float*)d_in[7];  const float* bq = (const float*)d_in[8];
  const float* Wo = (const float*)d_in[9];  const float* bo = (const float*)d_in[10];
  const float* gamma = (const float*)d_in[11];
  const float* beta  = (const float*)d_in[12];
  float* out = (float*)d_out;

  const size_t ACT = (size_t)ROWS * D_MODEL;          // 6,291,456 elems
  const size_t WEL = (size_t)D_MODEL * D_MODEL;       // 589,824 elems
  char* ws = (char*)d_ws;
  size_t off = 0;
  auto take = [&](size_t bytes) { char* p = ws + off; off += (bytes + 255) & ~(size_t)255; return p; };
  unsigned short* Xk  = (unsigned short*)take(ACT * 2);
  unsigned short* Xv  = (unsigned short*)take(ACT * 2);
  unsigned short* Xq  = (unsigned short*)take(ACT * 2);
  unsigned short* Wks = (unsigned short*)take(WEL * 2);
  unsigned short* Wvs = (unsigned short*)take(WEL * 2);
  unsigned short* Wqs = (unsigned short*)take(WEL * 2);
  unsigned short* Wos = (unsigned short*)take(WEL * 2);
  unsigned short* Qbf = (unsigned short*)take(ACT * 2);
  unsigned short* Kbf = (unsigned short*)take(ACT * 2);
  unsigned short* Vt  = (unsigned short*)take(ACT * 2);
  unsigned short* ctx = (unsigned short*)take(ACT * 2);

  {
    dim3 g((unsigned)((ACT + 255) / 256)), b(256);
    cvt_bf16_kernel<<<g, b, 0, stream>>>(key_in,   Xk, (int)ACT);
    cvt_bf16_kernel<<<g, b, 0, stream>>>(value_in, Xv, (int)ACT);
    cvt_bf16_kernel<<<g, b, 0, stream>>>(query_in, Xq, (int)ACT);
  }
  {
    dim3 g((unsigned)((WEL + 255) / 256)), b(256);
    swz_w_kernel<<<g, b, 0, stream>>>(Wk, Wks);
    swz_w_kernel<<<g, b, 0, stream>>>(Wv, Wvs);
    swz_w_kernel<<<g, b, 0, stream>>>(Wq, Wqs);
    swz_w_kernel<<<g, b, 0, stream>>>(Wo, Wos);
  }
  {
    dim3 g(ROWS / 16, D_MODEL / 96), b(32);
    proj_gemm_kernel<<<g, b, 0, stream>>>(Xk, Wks, bk, Kbf, 0);
    proj_gemm_kernel<<<g, b, 0, stream>>>(Xv, Wvs, bv, Vt,  1);
    proj_gemm_kernel<<<g, b, 0, stream>>>(Xq, Wqs, bq, Qbf, 2);
  }
  {
    dim3 g((BATCH * N_HEADS * SEQ / 16) / 8), b(256);
    flash_attn_kernel<<<g, b, 0, stream>>>(Qbf, Kbf, Vt, ctx);
  }
  {
    dim3 g(ROWS / 16), b(256);
    oproj_ln_kernel<<<g, b, 0, stream>>>(ctx, Wos, bo, query_in, gamma, beta, out);
  }
}
